// lstm_46428596469896
// MI455X (gfx1250) — matrix operand order repaired and checked by reading, not compiled, not run on hardware
//
#include <hip/hip_runtime.h>

// ---------------------------------------------------------------------------
// 3-layer LSTM (H=32) fused into one kernel, WMMA f32_16x16x32_f16 per step.
// One wave32 = one 16-batch tile. All h/c state stays on-chip for all T=512.
// K axis is permuted (sigma) so h write-back packs into b32 stores and the
// A operand is a contiguous 32B run per lane. B weights are pre-swizzled with
// the same permutation, so the WMMA contraction is unchanged.
// ---------------------------------------------------------------------------

typedef _Float16 f16;
typedef __attribute__((ext_vector_type(2)))  __fp16   v2fp;   // cvt_pkrtz result type
typedef __attribute__((ext_vector_type(8)))  _Float16 v8h;
typedef __attribute__((ext_vector_type(16))) _Float16 v16h;
typedef __attribute__((ext_vector_type(8)))  float    v8f;

#define WAVES 4        // waves per block (each wave = 16 batch rows)
#define TT    512      // timesteps
#define HH    32       // hidden
#define NT    8        // N tiles (gate dim 128 / 16)

// hardware-K -> storage position: swap bits 3 and 4
__host__ __device__ __forceinline__ int kpos(int K) {
    return (K & 7) | (((K >> 4) & 1) << 3) | (((K >> 3) & 1) << 4);
}
// storage position -> logical hidden unit
__host__ __device__ __forceinline__ int ulog(int p) {
    return (p >> 1) | ((p & 1) << 4);
}

__device__ __forceinline__ void lds_fence() {
    // order cross-lane LDS producer->consumer within this wave
    asm volatile("s_wait_dscnt 0" ::: "memory");
}
// fast sigmoid: v_mul + v_exp + v_add + v_rcp
__device__ __forceinline__ float sigf(float x) {
    return __builtin_amdgcn_rcpf(1.0f + __expf(-x));
}
// fast tanh: (1-e)/(1+e), e = exp(-2x) -> mul, exp, add, rcp, mul, fma
__device__ __forceinline__ float tanhx(float x) {
    float e = __expf(-2.0f * x);
    return fmaf(-2.0f, e * __builtin_amdgcn_rcpf(1.0f + e), 1.0f);
}

// A operand: with the sigma layout each lane reads 16 contiguous halfs:
// lanes 0-15 storage p=0..15 of row m=lane; lanes 16-31 p=16..31 of row m=lane-16
__device__ __forceinline__ v16h load_A(const f16* slot, int lane) {
    const int m = lane & 15;
    const f16* p = slot + m * HH + (lane >> 4) * 16;
    v8h lo = *(const v8h*)(p);
    v8h hi = *(const v8h*)(p + 8);
    return __builtin_shufflevector(lo, hi, 0,1,2,3,4,5,6,7,8,9,10,11,12,13,14,15);
}

// B operand (32x16 f16) pre-swizzled in LDS: 8 contiguous dwords per lane
__device__ __forceinline__ v16h load_B(const unsigned* wbuf, int mat, int n0, int lane) {
    const unsigned* p = wbuf + ((((mat * NT) + n0) * 32 + lane) << 3);
    v8h a = *(const v8h*)(p);
    v8h b = *(const v8h*)(p + 4);
    return __builtin_shufflevector(a, b, 0,1,2,3,4,5,6,7,8,9,10,11,12,13,14,15);
}

__device__ __forceinline__ v8f wmma_step(v16h A, v16h B, v8f C) {
    return __builtin_amdgcn_wmma_f32_16x16x32_f16(false, A, false, B, (short)0, C, false, false);
}

// LSTM cell update from D-layout accumulators; writes h (packed f16) to slot
__device__ __forceinline__ void gates_update(v8f* acc, float* cstL, f16* slot, int lane) {
    const int ncol = lane & 15;
    const int hi   = lane >> 4;
#pragma unroll
    for (int v = 0; v < 8; ++v) {          // rows: m = v + 8*hi
        float hv[2];
#pragma unroll
        for (int j = 0; j < 2; ++j) {      // hidden halves: u = 16*j + ncol
            float zi = acc[0 + j][v];
            float zf = acc[2 + j][v];
            float zg = acc[4 + j][v];
            float zo = acc[6 + j][v];
            float c  = sigf(zf) * cstL[j * 8 + v] + sigf(zi) * tanhx(zg);
            cstL[j * 8 + v] = c;
            hv[j] = sigf(zo) * tanhx(c);
        }
        // sigma(ncol)=2*ncol, sigma(ncol+16)=2*ncol+1 -> one packed b32 store
        v2fp pk = __builtin_amdgcn_cvt_pkrtz(hv[0], hv[1]);
        *(v2fp*)(slot + (v + 8 * hi) * HH + 2 * ncol) = pk;
    }
}

__global__ __launch_bounds__(WAVES * 32)
void lstm3_wmma_kernel(const float* __restrict__ x,
                       const float* __restrict__ Wih0, const float* __restrict__ Whh0,
                       const float* __restrict__ bih0, const float* __restrict__ bhh0,
                       const float* __restrict__ Wih1, const float* __restrict__ Whh1,
                       const float* __restrict__ bih1, const float* __restrict__ bhh1,
                       const float* __restrict__ Wih2, const float* __restrict__ Whh2,
                       const float* __restrict__ bih2, const float* __restrict__ bhh2,
                       const float* __restrict__ W2,   const float* __restrict__ b2,
                       float* __restrict__ out) {
    // weights in B-operand lane layout: [mat][n0][lane][8 dwords]  (5 * 8KB = 40KB)
    alignas(32) __shared__ unsigned wbuf[5 * NT * 32 * 8];
    // per-wave h buffers: 3 layers * [16 x 32] f16 (sigma column order)
    alignas(16) __shared__ f16 hbuf[WAVES * 3 * 16 * HH];
    alignas(16) __shared__ float xbuf[WAVES * 16];

    const int tid  = threadIdx.x;
    const int lane = tid & 31;
    const int wave = tid >> 5;

    // ---- stage weights into LDS in B-matrix register layout (f32 -> f16),
    //      applying the K permutation: hardware K reads logical col ulog(kpos(K))
    for (int idx = tid; idx < 5 * 2048; idx += blockDim.x) {
        const int mat = idx >> 11;
        const int r   = idx & 2047;
        const int n0  = r >> 8;
        const int ln  = (r >> 3) & 31;
        const int v   = r & 7;
        const int g   = n0 * 16 + (ln & 15);         // gate row 0..127
        const int k0  = (ln >> 4) * 16 + v * 2;      // storage position kpos(K) (B: lanes 0-15 p=0..15)
        const int c0  = ulog(k0);
        const int c1  = ulog(k0 + 1);
        const float* W;
        switch (mat) {
            case 0:  W = Whh0; break;
            case 1:  W = Wih1; break;
            case 2:  W = Whh1; break;
            case 3:  W = Wih2; break;
            default: W = Whh2; break;
        }
        union { f16 h[2]; unsigned u; } cv;
        cv.h[0] = (f16)W[g * HH + c0];
        cv.h[1] = (f16)W[g * HH + c1];
        wbuf[idx] = cv.u;
    }
    for (int i = tid; i < WAVES * 3 * 16 * HH; i += blockDim.x) hbuf[i] = (f16)0.0f;
    __syncthreads();

    const int b0   = (blockIdx.x * WAVES + wave) * 16;   // batch base for this wave
    f16*   hb      = &hbuf[wave * 3 * 16 * HH];
    float* xb      = &xbuf[wave * 16];
    const int ncol = lane & 15;

    // per-lane loop-invariant scalars: combined bias per N-tile, layer-0 Wih column
    float bias_r[3][NT], w0_r[NT];
#pragma unroll
    for (int n0 = 0; n0 < NT; ++n0) {
        const int g = n0 * 16 + ncol;
        bias_r[0][n0] = bih0[g] + bhh0[g];
        bias_r[1][n0] = bih1[g] + bhh1[g];
        bias_r[2][n0] = bih2[g] + bhh2[g];
        w0_r[n0]      = Wih0[g];            // [128 x 1]
    }

    float cst[3][16];                       // c state, D layout
#pragma unroll
    for (int l = 0; l < 3; ++l)
#pragma unroll
        for (int i = 0; i < 16; ++i) cst[l][i] = 0.0f;

    for (int t = 0; t < TT; ++t) {
        // ---- stage x[:,t] for this wave's batch tile ----
        if (lane < 16) xb[lane] = x[(size_t)(b0 + lane) * TT + t];
        lds_fence();                         // also covers prior-step h stores
        float xr[8];
        {
            const float* xp = xb + (lane >> 4) * 8;   // rows m = v + 8*hi
#pragma unroll
            for (int v = 0; v < 8; ++v) xr[v] = xp[v];
        }

        v8f acc[NT];

        // ================= layer 0: z = bias + x*Wih0 + h0 @ Whh0^T =========
#pragma unroll
        for (int n0 = 0; n0 < NT; ++n0) {
            v8f a;
#pragma unroll
            for (int v = 0; v < 8; ++v) a[v] = bias_r[0][n0] + w0_r[n0] * xr[v];
            acc[n0] = a;
        }
        {
            v16h A = load_A(hb, lane);
#pragma unroll
            for (int n0 = 0; n0 < NT; ++n0)
                acc[n0] = wmma_step(A, load_B(wbuf, 0, n0, lane), acc[n0]);
        }
        gates_update(acc, cst[0], hb, lane);

        // ================= layer 1 ==========================================
#pragma unroll
        for (int n0 = 0; n0 < NT; ++n0) {
            v8f a;
#pragma unroll
            for (int v = 0; v < 8; ++v) a[v] = bias_r[1][n0];
            acc[n0] = a;
        }
        lds_fence();                         // h0(t) stores -> visible
        {
            v16h Ain  = load_A(hb, lane);
            v16h Aown = load_A(hb + 16 * HH, lane);
#pragma unroll
            for (int n0 = 0; n0 < NT; ++n0) {
                acc[n0] = wmma_step(Ain,  load_B(wbuf, 1, n0, lane), acc[n0]);
                acc[n0] = wmma_step(Aown, load_B(wbuf, 2, n0, lane), acc[n0]);
            }
        }
        gates_update(acc, cst[1], hb + 16 * HH, lane);

        // ================= layer 2 ==========================================
#pragma unroll
        for (int n0 = 0; n0 < NT; ++n0) {
            v8f a;
#pragma unroll
            for (int v = 0; v < 8; ++v) a[v] = bias_r[2][n0];
            acc[n0] = a;
        }
        lds_fence();                         // h1(t) stores -> visible
        {
            v16h Ain  = load_A(hb + 16 * HH, lane);
            v16h Aown = load_A(hb + 2 * 16 * HH, lane);
#pragma unroll
            for (int n0 = 0; n0 < NT; ++n0) {
                acc[n0] = wmma_step(Ain,  load_B(wbuf, 3, n0, lane), acc[n0]);
                acc[n0] = wmma_step(Aown, load_B(wbuf, 4, n0, lane), acc[n0]);
            }
        }
        gates_update(acc, cst[2], hb + 2 * 16 * HH, lane);
    }

    // ---- final projection: out[b0+m, :] = h2_last[m] @ W2^T + b2 ----
    lds_fence();
    {
        const int m  = ncol;
        const int ob = (lane >> 4) * 15;     // lanes 0-15: o=0..14, lanes 16-31: o=15..29
        const f16* h2 = hb + 2 * 16 * HH + m * HH;
        for (int oo = 0; oo < 15; ++oo) {
            const int o = ob + oo;
            float s = b2[o];
#pragma unroll
            for (int k = 0; k < HH; ++k) {
                const int sp = 2 * (k & 15) + (k >> 4);   // sigma(k)
                s += (float)h2[sp] * W2[o * HH + k];
            }
            out[(size_t)(b0 + m) * 30 + o] = s;
        }
    }
}

extern "C" void kernel_launch(void* const* d_in, const int* in_sizes, int n_in,
                              void* d_out, int out_size, void* d_ws, size_t ws_size,
                              hipStream_t stream) {
    const float* x    = (const float*)d_in[0];
    const float* Wih0 = (const float*)d_in[1];
    const float* Whh0 = (const float*)d_in[2];
    const float* bih0 = (const float*)d_in[3];
    const float* bhh0 = (const float*)d_in[4];
    const float* Wih1 = (const float*)d_in[5];
    const float* Whh1 = (const float*)d_in[6];
    const float* bih1 = (const float*)d_in[7];
    const float* bhh1 = (const float*)d_in[8];
    const float* Wih2 = (const float*)d_in[9];
    const float* Whh2 = (const float*)d_in[10];
    const float* bih2 = (const float*)d_in[11];
    const float* bhh2 = (const float*)d_in[12];
    const float* W2   = (const float*)d_in[13];
    const float* b2   = (const float*)d_in[14];
    float* out = (float*)d_out;

    const int B = in_sizes[0] / TT;            // 4096
    dim3 block(WAVES * 32);                    // 128 threads = 4 waves
    dim3 grid(B / (WAVES * 16));               // 64 blocks

    lstm3_wmma_kernel<<<grid, block, 0, stream>>>(
        x, Wih0, Whh0, bih0, bhh0, Wih1, Whh1, bih1, bhh1,
        Wih2, Whh2, bih2, bhh2, W2, b2, out);
}
